// VectorQuantizer_68161130988016
// MI455X (gfx1250) — compile-verified, matrix operand order checked by reading, not run on hardware
//
#include <hip/hip_runtime.h>
#include <hip/hip_bf16.h>

// Problem constants (from reference): B=64, T=1024, D=256, K=1024
#define K_CODES 1024
#define DIM     256
#define N_TOK   65536          // B*T
#define DECAYC  0.99f
#define BETAC   0.25f
#define EPSC    1e-5f

typedef __attribute__((ext_vector_type(16))) __bf16 v16bf;
typedef __attribute__((ext_vector_type(8)))  __bf16 v8bf;
typedef __attribute__((ext_vector_type(8)))  float  v8f;

// ---------------------------------------------------------------------------
// Prep: convert codebook to bf16, zero scratch accumulators.
// ---------------------------------------------------------------------------
__global__ void vq_prep(const float* __restrict__ embed,
                        __bf16* __restrict__ ebf,
                        float* __restrict__ dw,
                        float* __restrict__ counts,
                        float* __restrict__ loss_acc,
                        float* __restrict__ n_acc) {
    int i = blockIdx.x * blockDim.x + threadIdx.x;
    if (i < K_CODES * DIM) {
        ebf[i] = (__bf16)embed[i];
        dw[i]  = 0.0f;
    }
    if (i < K_CODES) counts[i] = 0.0f;
    if (i == 0) { *loss_acc = 0.0f; *n_acc = 0.0f; }
}

// 0.5 * ||e_k||^2 in fp32 (argmin(dist) == argmax(z.e - 0.5||e||^2))
__global__ void vq_norms(const float* __restrict__ embed,
                         float* __restrict__ halfn) {
    int k = blockIdx.x * blockDim.x + threadIdx.x;
    if (k >= K_CODES) return;
    const float* row = embed + (size_t)k * DIM;
    float s = 0.0f;
    for (int d = 0; d < DIM; ++d) { float v = row[d]; s += v * v; }
    halfn[k] = 0.5f * s;
}

// ---------------------------------------------------------------------------
// Main: one wave32 owns 16 tokens. bf16 WMMA distance GEMM + argmax,
// fp32 gather / loss / scatter.
// ---------------------------------------------------------------------------
__global__ __launch_bounds__(256)
void vq_main(const float*  __restrict__ z,
             const float*  __restrict__ embed,
             const __bf16* __restrict__ ebf,
             const float*  __restrict__ halfn,
             float* __restrict__ dw,
             float* __restrict__ counts,
             float* __restrict__ loss_acc,
             float* __restrict__ q_out,
             float* __restrict__ idx_out) {
    const int lane   = threadIdx.x & 31;
    const int wave   = blockIdx.x * (blockDim.x >> 5) + (threadIdx.x >> 5);
    const int tokenBase = wave * 16;
    const int lane15 = lane & 15;
    const int half   = lane >> 4;

    // ---- Load 16x256 z tile once, convert to bf16 A-fragments --------------
    // 16-bit A 16x32 layout: lane = M row; K chunks at (lane>>4)*8 and +16.
    const float* zr = z + (size_t)(tokenBase + lane15) * DIM;
    v16bf a[8];
#pragma unroll
    for (int kt = 0; kt < 8; ++kt) {
        const int d0 = kt * 32 + half * 8;
        const float4 f0 = *(const float4*)(zr + d0);
        const float4 f1 = *(const float4*)(zr + d0 + 4);
        const float4 f2 = *(const float4*)(zr + d0 + 16);
        const float4 f3 = *(const float4*)(zr + d0 + 20);
        v16bf t;
        t[0]  = (__bf16)f0.x; t[1]  = (__bf16)f0.y; t[2]  = (__bf16)f0.z; t[3]  = (__bf16)f0.w;
        t[4]  = (__bf16)f1.x; t[5]  = (__bf16)f1.y; t[6]  = (__bf16)f1.z; t[7]  = (__bf16)f1.w;
        t[8]  = (__bf16)f2.x; t[9]  = (__bf16)f2.y; t[10] = (__bf16)f2.z; t[11] = (__bf16)f2.w;
        t[12] = (__bf16)f3.x; t[13] = (__bf16)f3.y; t[14] = (__bf16)f3.z; t[15] = (__bf16)f3.w;
        a[kt] = t;
    }

    float bestV[8];
    int   bestI[8];
#pragma unroll
    for (int r = 0; r < 8; ++r) { bestV[r] = -3.4e38f; bestI[r] = 0; }

    // ---- 64 code tiles x 8 k-steps of v_wmma_f32_16x16x32_bf16 -------------
    for (int nt = 0; nt < 64; ++nt) {
        // B 32x16 layout mirrors A: lane = N column = codebook row.
        const __bf16* er = ebf + (size_t)(nt * 16 + lane15) * DIM;
        v8f acc = {};
#pragma unroll
        for (int kt = 0; kt < 8; ++kt) {
            const int d0 = kt * 32 + half * 8;
            const v8bf b0 = *(const v8bf*)(er + d0);
            const v8bf b1 = *(const v8bf*)(er + d0 + 16);
            v16bf b;
#pragma unroll
            for (int j = 0; j < 8; ++j) { b[j] = b0[j]; b[8 + j] = b1[j]; }
            acc = __builtin_amdgcn_wmma_f32_16x16x32_bf16(
                false, a[kt], false, b, (short)0, acc, false, false);
        }
        const float hn = halfn[nt * 16 + lane15];
        const int   c  = nt * 16 + lane15;
#pragma unroll
        for (int r = 0; r < 8; ++r) {
            const float s = acc[r] - hn;            // argmax(z.e - 0.5||e||^2)
            if (s > bestV[r]) { bestV[r] = s; bestI[r] = c; }
        }
    }

    // ---- Cross-lane argmax within each 16-lane half (C: lane = N) ----------
#pragma unroll
    for (int mask = 1; mask <= 8; mask <<= 1) {
#pragma unroll
        for (int r = 0; r < 8; ++r) {
            const float ov = __shfl_xor(bestV[r], mask, 32);
            const int   oi = __shfl_xor(bestI[r], mask, 32);
            if (ov > bestV[r] || (ov == bestV[r] && oi < bestI[r])) {
                bestV[r] = ov; bestI[r] = oi;
            }
        }
    }

    // idx output: lane0 -> rows 0..7, lane16 -> rows 8..15
    if (lane15 == 0) {
#pragma unroll
        for (int r = 0; r < 8; ++r)
            idx_out[tokenBase + half * 8 + r] = (float)bestI[r];
    }

    // ---- Gather q (fp32), q_st output, loss, EMA scatter --------------------
    float lsum = 0.0f;
#pragma unroll
    for (int t = 0; t < 16; ++t) {
        const int code = __shfl(bestI[t & 7], (t < 8) ? 0 : 16, 32);
        const int row  = tokenBase + t;
        const int d    = lane * 8;                       // 32 lanes x 8 dims
        const float4 z0 = *(const float4*)(z + (size_t)row * DIM + d);
        const float4 z1 = *(const float4*)(z + (size_t)row * DIM + d + 4);
        const float4 e0 = *(const float4*)(embed + (size_t)code * DIM + d);
        const float4 e1 = *(const float4*)(embed + (size_t)code * DIM + d + 4);
        *(float4*)(q_out + (size_t)row * DIM + d)     = e0;   // q_st == q
        *(float4*)(q_out + (size_t)row * DIM + d + 4) = e1;
        float dx;
        dx = z0.x - e0.x; lsum += dx * dx;
        dx = z0.y - e0.y; lsum += dx * dx;
        dx = z0.z - e0.z; lsum += dx * dx;
        dx = z0.w - e0.w; lsum += dx * dx;
        dx = z1.x - e1.x; lsum += dx * dx;
        dx = z1.y - e1.y; lsum += dx * dx;
        dx = z1.z - e1.z; lsum += dx * dx;
        dx = z1.w - e1.w; lsum += dx * dx;
        float* dp = dw + (size_t)code * DIM + d;              // dw[idx] += z
        atomicAdd(dp + 0, z0.x); atomicAdd(dp + 1, z0.y);
        atomicAdd(dp + 2, z0.z); atomicAdd(dp + 3, z0.w);
        atomicAdd(dp + 4, z1.x); atomicAdd(dp + 5, z1.y);
        atomicAdd(dp + 6, z1.z); atomicAdd(dp + 7, z1.w);
        if (lane == 0) atomicAdd(&counts[code], 1.0f);
    }

#pragma unroll
    for (int mask = 16; mask >= 1; mask >>= 1) lsum += __shfl_xor(lsum, mask, 32);
    if (lane == 0) atomicAdd(loss_acc, lsum);
}

// ---------------------------------------------------------------------------
// Finalize: EMA counts + global sum, then smoothed embed / ema_w / loss.
// ---------------------------------------------------------------------------
__global__ void vq_count(const float* __restrict__ ema_count,
                         const float* __restrict__ counts,
                         float* __restrict__ count_out,
                         float* __restrict__ n_acc) {
    int k = blockIdx.x * blockDim.x + threadIdx.x;
    if (k >= K_CODES) return;
    const float nc = DECAYC * ema_count[k] + (1.0f - DECAYC) * counts[k];
    count_out[k] = nc;
    atomicAdd(n_acc, nc);
}

__global__ void vq_final(const float* __restrict__ ema_weight,
                         const float* __restrict__ dw,
                         const float* __restrict__ count_out,
                         const float* __restrict__ n_acc,
                         const float* __restrict__ loss_acc,
                         float* __restrict__ embed_out,
                         float* __restrict__ emaw_out,
                         float* __restrict__ loss_out) {
    int i = blockIdx.x * blockDim.x + threadIdx.x;
    if (i >= K_CODES * DIM) return;
    const int   k  = i >> 8;                 // DIM == 256
    const float n  = *n_acc;
    const float nc = count_out[k];
    const float cs = (nc + EPSC) / (n + (float)K_CODES * EPSC) * n;
    const float w  = DECAYC * ema_weight[i] + (1.0f - DECAYC) * dw[i];
    emaw_out[i]  = w;
    embed_out[i] = w / cs;
    if (i == 0)                              // commit == codebook numerically
        *loss_out = (1.0f + BETAC) * (*loss_acc) / (float)((size_t)N_TOK * DIM);
}

// ---------------------------------------------------------------------------
extern "C" void kernel_launch(void* const* d_in, const int* in_sizes, int n_in,
                              void* d_out, int out_size, void* d_ws, size_t ws_size,
                              hipStream_t stream) {
    const float* z          = (const float*)d_in[0];   // [B,T,D]
    const float* embed      = (const float*)d_in[1];   // [K,D]
    const float* ema_count  = (const float*)d_in[2];   // [K]
    const float* ema_weight = (const float*)d_in[3];   // [K,D]

    // Outputs, concatenated flat in reference return order.
    float* out       = (float*)d_out;
    float* q_out     = out;                                   // B*T*D
    float* loss_out  = out + (size_t)N_TOK * DIM;             // 1
    float* idx_out   = loss_out + 1;                          // B*T
    float* embed_out = idx_out + N_TOK;                       // K*D
    float* count_out = embed_out + (size_t)K_CODES * DIM;     // K
    float* emaw_out  = count_out + K_CODES;                   // K*D

    // Scratch layout (~1.6 MB): dw | counts | halfn | loss | n | pad | ebf
    float*  dw       = (float*)d_ws;                          // K*D
    float*  counts   = dw + (size_t)K_CODES * DIM;            // K
    float*  halfn    = counts + K_CODES;                      // K
    float*  loss_acc = halfn + K_CODES;                       // 1
    float*  n_acc    = loss_acc + 1;                          // 1
    __bf16* ebf      = (__bf16*)(n_acc + 3);                  // 16B-aligned, K*D bf16

    vq_prep <<<(K_CODES * DIM + 255) / 256, 256, 0, stream>>>(embed, ebf, dw, counts, loss_acc, n_acc);
    vq_norms<<<(K_CODES + 255) / 256,       256, 0, stream>>>(embed, halfn);
    // 4096 waves (N_TOK/16), 8 waves per 256-thread block -> 512 blocks
    vq_main <<<(N_TOK / 16) / 8,            256, 0, stream>>>(z, embed, ebf, halfn, dw, counts,
                                                              loss_acc, q_out, idx_out);
    vq_count<<<(K_CODES + 255) / 256,       256, 0, stream>>>(ema_count, counts, count_out, n_acc);
    vq_final<<<(K_CODES * DIM + 255) / 256, 256, 0, stream>>>(ema_weight, dw, count_out, n_acc,
                                                              loss_acc, embed_out, emaw_out, loss_out);
}
